// DGCNN_37881611551020
// MI455X (gfx1250) — matrix operand order repaired and checked by reading, not compiled, not run on hardware
//
#include <hip/hip_runtime.h>
#include <hip/hip_bf16.h>

typedef __attribute__((ext_vector_type(16))) _Float16 v16h;
typedef __attribute__((ext_vector_type(8)))  _Float16 v8h;
typedef __attribute__((ext_vector_type(8)))  float    v8f;

#define NB 8          // batch
#define KNNK 32       // k-nearest neighbors
#define GEMM_WAVES 8  // 256-thread blocks = 8 wave32

// ---------------------------------------------------------------------------
// KNN: one thread per query point, brute-force top-32 by insertion.
// Ordering among the 32 is irrelevant downstream (max-pool / BN stats).
// ---------------------------------------------------------------------------
__global__ __launch_bounds__(128)
void knn_kernel(const float* __restrict__ pcd, int* __restrict__ knn,
                int npq, int npp) {
  int idx = blockIdx.x * blockDim.x + threadIdx.x;
  if (idx >= NB * npq) return;
  int b = idx / npq, q = idx - b * npq;
  const float* base = pcd + (size_t)b * 4096 * 3;
  float qx = base[q * 3 + 0], qy = base[q * 3 + 1], qz = base[q * 3 + 2];
  float bd[KNNK];
  int   bi[KNNK];
  #pragma unroll
  for (int i = 0; i < KNNK; ++i) { bd[i] = 3.4e38f; bi[i] = 0; }
  for (int j = 0; j < npp; ++j) {
    float dx = base[j * 3 + 0] - qx;
    float dy = base[j * 3 + 1] - qy;
    float dz = base[j * 3 + 2] - qz;
    float d = dx * dx + dy * dy + dz * dz;
    if (d < bd[KNNK - 1]) {
      int pos = KNNK - 1;
      while (pos > 0 && bd[pos - 1] > d) {
        bd[pos] = bd[pos - 1]; bi[pos] = bi[pos - 1]; --pos;
      }
      bd[pos] = d; bi[pos] = j;
    }
  }
  int* o = knn + (size_t)idx * KNNK;
  #pragma unroll
  for (int i = 0; i < KNNK; ++i) o[i] = bi[i];
}

// ---------------------------------------------------------------------------
// Pack f32 weights W[cout][cinAct] into f16 WMMA B-operand lane layout:
// per (colTile, kChunk): 32 lanes x 16 halves. contiguous k order (same as A):
// lanes 0-15 hold K=0..15 (column n = lane), lanes 16-31 hold K=16..31.
// ---------------------------------------------------------------------------
__global__ __launch_bounds__(256)
void packw_kernel(const float* __restrict__ W, _Float16* __restrict__ Bp,
                  int colTiles, int chunks, int cinAct) {
  int gid = blockIdx.x * blockDim.x + threadIdx.x;
  int total = colTiles * chunks * 32;
  if (gid >= total) return;
  int lane = gid & 31;
  int kc   = (gid >> 5) % chunks;
  int tile = gid / (32 * chunks);
  int n    = lane & 15;
  int khi  = lane >> 4;
  _Float16* dst = Bp + (size_t)gid * 16;
  #pragma unroll
  for (int j = 0; j < 16; ++j) {
    int k = kc * 32 + khi * 16 + j;
    float v = (k < cinAct) ? W[(size_t)(tile * 16 + n) * cinAct + k] : 0.f;
    dst[j] = (_Float16)v;
  }
}

// ---------------------------------------------------------------------------
// WMMA GEMM: rows x cout = A(rows x cinP) * B(cinP x cout) + bias.
// mode 0: A built on the fly from the edge feature (gathered from f via knn):
//         E[row][k] = k<fdim   ? f[b][nbr][k] - f[b][m][k]
//                   : k<2*fdim ? f[b][m][k-fdim]   : 0 (zero K-padding)
// mode 1: A[row][k] = relu(h_in[row][k]*scale[k] + shift[k])  (BN fused)
// Block = 16 rows x all cout. Key points:
//  * all 16 rows of a block share (batch, point): tiles are 16-aligned inside
//    the 32-neighbor groups -> decode once (block-uniform scalars).
//  * B tiles are loaded BEFORE the A LDS fill + barrier so their global
//    latency overlaps staging instead of stalling right at the v_wmma.
// ---------------------------------------------------------------------------
__global__ __launch_bounds__(256)
void gemm_wmma_kernel(const float* __restrict__ fsrc, const int* __restrict__ knn,
                      const float* __restrict__ hin,  const float* __restrict__ bnp,
                      const _Float16* __restrict__ Bp,
                      const float* __restrict__ bias, float* __restrict__ hout,
                      int cout, int cinP, int cinAct, int fdim,
                      int npq, int npp, int mode) {
  __shared__ __align__(16) _Float16 As[16 * 32];
  const int row0 = blockIdx.x * 16;
  const int lane = threadIdx.x & 31;
  const int wave = threadIdx.x >> 5;
  const int colTiles = cout >> 4;
  const int chunks = cinP >> 5;
  const int mrow = lane & 15;
  const int khalf = lane >> 4;
  const bool hasT0 = (wave < colTiles);
  const bool hasT1 = (wave + GEMM_WAVES < colTiles);

  // block-uniform row decode (all 16 rows share b and mp)
  const float* fq = nullptr;   // f[b][mp][.]
  const float* fbase = nullptr; // f[b][0][.]
  const float* hrow = nullptr;  // hin[row0][.]
  if (mode == 0) {
    int ub  = row0 / (npq * KNNK);
    int rem = row0 - ub * npq * KNNK;
    int ump = rem >> 5;
    fbase = fsrc + (size_t)ub * npp * fdim;
    fq    = fbase + (size_t)ump * fdim;
  } else {
    hrow = hin + (size_t)row0 * cinAct;
  }

  const v16h* bp = (const v16h*)Bp;
  v8f acc0 = {}; v8f acc1 = {};

  for (int kc = 0; kc < chunks; ++kc) {
    // ---- prefetch B tiles for this chunk (overlaps A staging below) ----
    v16h bm0 = {};
    v16h bm1 = {};
    if (hasT0) bm0 = bp[((size_t)wave * chunks + kc) * 32 + lane];
    if (hasT1) bm1 = bp[((size_t)(wave + GEMM_WAVES) * chunks + kc) * 32 + lane];

    // ---- stage A tile (16 rows x 32 K) into LDS as f16 ----
    for (int e = threadIdx.x; e < 512; e += 256) {
      int m = e >> 5, kl = e & 31;
      int kg = (kc << 5) + kl;
      float v = 0.f;
      if (mode == 0) {
        if (kg < fdim) {
          int nbr = knn[row0 + m];
          v = fbase[(size_t)nbr * fdim + kg] - fq[kg];
        } else if (kg < 2 * fdim) {
          v = fq[kg - fdim];
        }
      } else {
        if (kg < cinAct) {
          float x = hrow[(size_t)m * cinAct + kg];
          v = fmaxf(x * bnp[kg] + bnp[cinAct + kg], 0.f);
        }
      }
      As[(m << 5) + kl] = (_Float16)v;
    }
    __syncthreads();

    // ---- build A operand per lane (contiguous k order, same as packed B) ----
    // lane<16: K {0..15}; lane>=16: K {16..31}; row M=lane%16
    const v8h* arow = (const v8h*)(As + (mrow << 5));
    v8h alo = arow[2 * khalf];
    v8h ahi = arow[2 * khalf + 1];
    v16h a;
    #pragma unroll
    for (int j = 0; j < 8; ++j) { a[j] = alo[j]; a[8 + j] = ahi[j]; }

    if (hasT0)
      acc0 = __builtin_amdgcn_wmma_f32_16x16x32_f16(
          false, a, false, bm0, (short)0, acc0, false, false);
    if (hasT1)
      acc1 = __builtin_amdgcn_wmma_f32_16x16x32_f16(
          false, a, false, bm1, (short)0, acc1, false, false);
    __syncthreads();
  }

  // ---- store D + bias (C/D layout: VGPR r holds M=r / M=r+8) ----
  #pragma unroll
  for (int ti = 0; ti < 2; ++ti) {
    int t = wave + ti * GEMM_WAVES;
    if (t < colTiles) {
      int c = (t << 4) + mrow;
      float bv = bias[c];
      v8f acc = (ti == 0) ? acc0 : acc1;
      #pragma unroll
      for (int r = 0; r < 8; ++r) {
        int m = r + (khalf << 3);
        hout[(size_t)(row0 + m) * cout + c] = acc[r] + bv;
      }
    }
  }
}

// ---------------------------------------------------------------------------
// BN stats pass: out[0..C) += sum, out[C..2C) += sumsq.
// C | 256 => each thread owns a fixed channel (t % C) across its stride loop.
// ---------------------------------------------------------------------------
__global__ __launch_bounds__(256)
void stats_kernel(const float* __restrict__ h, float* __restrict__ out,
                  size_t total, int C) {
  __shared__ float s[512];
  int t = threadIdx.x;
  float sum = 0.f, sq = 0.f;
  size_t stride = (size_t)blockDim.x * gridDim.x;
  for (size_t i = (size_t)blockIdx.x * blockDim.x + t; i < total; i += stride) {
    float v = h[i]; sum += v; sq += v * v;
  }
  s[t] = sum; s[256 + t] = sq;
  __syncthreads();
  if (t < C) {
    float S = 0.f, Q = 0.f;
    for (int j = t; j < 256; j += C) { S += s[j]; Q += s[256 + j]; }
    atomicAdd(&out[t], S);
    atomicAdd(&out[C + t], Q);
  }
}

__global__ void zero_kernel(float* p, int n) {
  int i = blockIdx.x * blockDim.x + threadIdx.x;
  if (i < n) p[i] = 0.f;
}

// stats -> (scale, shift): bn(x) = x*scale + shift
__global__ void bnfinal_kernel(const float* __restrict__ raw,
                               const float* __restrict__ g,
                               const float* __restrict__ be,
                               float* __restrict__ bnp, int C, float invN) {
  int c = blockIdx.x * blockDim.x + threadIdx.x;
  if (c >= C) return;
  float mean = raw[c] * invN;
  float var  = raw[C + c] * invN - mean * mean;
  float sc = g[c] * rsqrtf(var + 1e-5f);
  bnp[c] = sc;
  bnp[C + c] = be[c] - mean * sc;
}

// ---------------------------------------------------------------------------
// feat[b][c][m] = max_k relu(h2[row][c]*scale + shift), rows = (b*npq+m)*32+k
// ---------------------------------------------------------------------------
__global__ __launch_bounds__(256)
void maxpool_kernel(const float* __restrict__ h2, const float* __restrict__ bnp,
                    float* __restrict__ feat, int npq, int C) {
  int idx = blockIdx.x * blockDim.x + threadIdx.x;
  int total = NB * npq * C;
  if (idx >= total) return;
  int c = idx % C;
  int pm = idx / C;
  float sc = bnp[c], sh = bnp[C + c];
  const float* p = h2 + (size_t)pm * KNNK * C + c;
  float mx = 0.f;   // relu output >= 0
  #pragma unroll 4
  for (int kk = 0; kk < KNNK; ++kk) {
    float v = fmaxf(p[(size_t)kk * C] * sc + sh, 0.f);
    mx = fmaxf(mx, v);
  }
  int b = pm / npq, m = pm - b * npq;
  feat[((size_t)b * C + c) * npq + m] = mx;
}

// f[b][m][:] = concat(feat[b][:,m], xyz[b][m]); xyz is always a prefix of pcd
__global__ __launch_bounds__(256)
void fbuild_kernel(const float* __restrict__ feat, const float* __restrict__ pcd,
                   float* __restrict__ f, int npq, int C, int fdim) {
  int idx = blockIdx.x * blockDim.x + threadIdx.x;
  int total = NB * npq * fdim;
  if (idx >= total) return;
  int j = idx % fdim;
  int pm = idx / fdim;
  int b = pm / npq, m = pm - b * npq;
  float v;
  if (j < C) v = feat[((size_t)b * C + j) * npq + m];
  else       v = pcd[((size_t)b * 4096 + m) * 3 + (j - C)];
  f[idx] = v;
}

// out[b][o][n] = fcb[o] + sum_c fcW[o][c] * feat[b][c][n]
__global__ __launch_bounds__(256)
void fc_kernel(const float* __restrict__ feat, const float* __restrict__ W,
               const float* __restrict__ bvec, float* __restrict__ out,
               int N, int Cin, int Cout) {
  int idx = blockIdx.x * blockDim.x + threadIdx.x;
  int total = NB * Cout * N;
  if (idx >= total) return;
  int n = idx % N;
  int t = idx / N;
  int o = t % Cout;
  int b = t / Cout;
  const float* fb = feat + (size_t)b * Cin * N + n;
  const float* w  = W + (size_t)o * Cin;
  float acc = bvec[o];
  for (int c = 0; c < Cin; ++c) acc += w[c] * fb[(size_t)c * N];
  out[idx] = acc;
}

__global__ void xyzout_kernel(const float* __restrict__ pcd, float* __restrict__ out) {
  int i = blockIdx.x * blockDim.x + threadIdx.x;
  if (i >= NB * 1024 * 3) return;
  int b = i / (1024 * 3), r = i - b * (1024 * 3);
  out[i] = pcd[(size_t)b * 4096 * 3 + r];
}

__global__ void idxout_kernel(int* __restrict__ out) {
  int i = blockIdx.x * blockDim.x + threadIdx.x;
  if (i >= NB * 1024) return;
  out[i] = i & 1023;
}

__global__ void copyf_kernel(const float* __restrict__ src, float* __restrict__ dst, int n) {
  int i = blockIdx.x * blockDim.x + threadIdx.x;
  if (i < n) dst[i] = src[i];
}

// ---------------------------------------------------------------------------
extern "C" void kernel_launch(void* const* d_in, const int* in_sizes, int n_in,
                              void* d_out, int out_size, void* d_ws, size_t ws_size,
                              hipStream_t stream) {
  (void)in_sizes; (void)n_in; (void)out_size; (void)ws_size;
  const float* pcd = (const float*)d_in[0];
  const float* fcW = (const float*)d_in[25];
  const float* fcb = (const float*)d_in[26];

  // ---- workspace bump allocator ----
  char* ws = (char*)d_ws;
  size_t off = 0;
  auto walloc = [&](size_t bytes) -> void* {
    void* p = ws + off;
    off = (off + bytes + 255) & ~(size_t)255;
    return p;
  };
  int*      knnb  = (int*)     walloc(sizeof(int)      * (size_t)NB * 4096 * KNNK);
  float*    f2    = (float*)   walloc(sizeof(float)    * (size_t)NB * 4096 * 67);
  float*    f3    = (float*)   walloc(sizeof(float)    * (size_t)NB * 2048 * 131);
  float*    hA    = (float*)   walloc(sizeof(float)    * (size_t)NB * 4096 * KNNK * 64);
  float*    hB    = (float*)   walloc(sizeof(float)    * (size_t)NB * 4096 * KNNK * 64);
  float*    feat1 = (float*)   walloc(sizeof(float)    * (size_t)NB * 64  * 4096);
  float*    feat2 = (float*)   walloc(sizeof(float)    * (size_t)NB * 128 * 2048);
  float*    feat3 = (float*)   walloc(sizeof(float)    * (size_t)NB * 256 * 1024);
  _Float16* BpA   = (_Float16*)walloc(sizeof(_Float16) * 256 * 288);
  _Float16* BpB   = (_Float16*)walloc(sizeof(_Float16) * 256 * 256);
  float*    rawA  = (float*)   walloc(sizeof(float) * 512);
  float*    bnpA  = (float*)   walloc(sizeof(float) * 512);
  float*    rawB  = (float*)   walloc(sizeof(float) * 512);
  float*    bnpB  = (float*)   walloc(sizeof(float) * 512);

  struct LCfg { int npq, npp, fdim, cin, cinP, cmid, cout; const float* f; float* feat; };
  const LCfg L[3] = {
    {4096, 4096,   3,   6,  32,  64,  64, pcd, feat1},
    {2048, 4096,  67, 134, 160, 128, 128, f2,  feat2},
    {1024, 2048, 131, 262, 288, 256, 256, f3,  feat3},
  };
  float* fnext[3] = { f2, f3, nullptr };

  for (int li = 0; li < 3; ++li) {
    const LCfg& cf = L[li];
    int base = 1 + li * 8;
    const float* Wa  = (const float*)d_in[base + 0];
    const float* ba  = (const float*)d_in[base + 1];
    const float* ga  = (const float*)d_in[base + 2];
    const float* bea = (const float*)d_in[base + 3];
    const float* Wb  = (const float*)d_in[base + 4];
    const float* bb  = (const float*)d_in[base + 5];
    const float* gb  = (const float*)d_in[base + 6];
    const float* beb = (const float*)d_in[base + 7];
    int rows = NB * cf.npq * KNNK;
    float invN = 1.0f / (float)rows;

    // KNN
    knn_kernel<<<(NB * cf.npq + 127) / 128, 128, 0, stream>>>(pcd, knnb, cf.npq, cf.npp);

    // conv1: edge-gather GEMM (A on the fly), D = E*Wa^T + ba
    int ct1 = cf.cmid / 16, ch1 = cf.cinP / 32;
    packw_kernel<<<(ct1 * ch1 * 32 + 255) / 256, 256, 0, stream>>>(Wa, BpA, ct1, ch1, cf.cin);
    gemm_wmma_kernel<<<rows / 16, 256, 0, stream>>>(
        cf.f, knnb, nullptr, nullptr, BpA, ba, hA,
        cf.cmid, cf.cinP, cf.cin, cf.fdim, cf.npq, cf.npp, 0);

    // BN1 stats
    zero_kernel<<<1, 512, 0, stream>>>(rawA, 2 * cf.cmid);
    stats_kernel<<<512, 256, 0, stream>>>(hA, rawA, (size_t)rows * cf.cmid, cf.cmid);
    bnfinal_kernel<<<1, 256, 0, stream>>>(rawA, ga, bea, bnpA, cf.cmid, invN);

    // conv2: A = relu(bn(hA)) fused into GEMM A-load
    int ct2 = cf.cout / 16, ch2 = cf.cmid / 32;
    packw_kernel<<<(ct2 * ch2 * 32 + 255) / 256, 256, 0, stream>>>(Wb, BpB, ct2, ch2, cf.cmid);
    gemm_wmma_kernel<<<rows / 16, 256, 0, stream>>>(
        nullptr, nullptr, hA, bnpA, BpB, bb, hB,
        cf.cout, cf.cmid, cf.cmid, 0, cf.npq, cf.npp, 1);

    // BN2 stats
    zero_kernel<<<1, 512, 0, stream>>>(rawB, 2 * cf.cout);
    stats_kernel<<<512, 256, 0, stream>>>(hB, rawB, (size_t)rows * cf.cout, cf.cout);
    bnfinal_kernel<<<1, 256, 0, stream>>>(rawB, gb, beb, bnpB, cf.cout, invN);

    // relu(bn) + max over k -> feat[b][c][m]
    int tp = NB * cf.npq * cf.cout;
    maxpool_kernel<<<(tp + 255) / 256, 256, 0, stream>>>(hB, bnpB, cf.feat, cf.npq, cf.cout);

    // build f for next layer: concat(feat^T, xyz-prefix)
    if (li < 2) {
      int fdim2 = cf.cout + 3;
      int tf = NB * cf.npq * fdim2;
      fbuild_kernel<<<(tf + 255) / 256, 256, 0, stream>>>(cf.feat, pcd, fnext[li], cf.npq, cf.cout, fdim2);
    }
  }

  // ---- outputs: [xyz | out_feat | idx | feat] flat ----
  float* out = (float*)d_out;
  const size_t off_xyz   = 0;
  const size_t off_feat2 = (size_t)NB * 1024 * 3;                 // out_feat
  const size_t off_idx   = off_feat2 + (size_t)NB * 128 * 1024;
  const size_t off_f3    = off_idx + (size_t)NB * 1024;

  xyzout_kernel<<<(NB * 1024 * 3 + 255) / 256, 256, 0, stream>>>(pcd, out + off_xyz);
  int totFC = NB * 128 * 1024;
  fc_kernel<<<(totFC + 255) / 256, 256, 0, stream>>>(feat3, fcW, fcb, out + off_feat2, 1024, 256, 128);
  idxout_kernel<<<(NB * 1024 + 255) / 256, 256, 0, stream>>>((int*)(out + off_idx));
  int nf3 = NB * 256 * 1024;
  copyf_kernel<<<(nf3 + 255) / 256, 256, 0, stream>>>(feat3, out + off_f3, nf3);
}
